// SFRGNNSegmentor_85323820302935
// MI455X (gfx1250) — matrix operand order repaired and checked by reading, not compiled, not run on hardware
//
#include <hip/hip_runtime.h>
#include <hip/hip_bf16.h>

typedef __attribute__((ext_vector_type(16))) __bf16 v16bf;
typedef __attribute__((ext_vector_type(8)))  __bf16 v8bf;
typedef __attribute__((ext_vector_type(8)))  float  v8f;

#define N_NODES  50000
#define N_EDGES  400000
#define N_GRAPHS 250

__device__ __forceinline__ float mishf(float x) {
    float sp = (x > 20.f) ? x : log1pf(__expf(x));
    return x * tanhf(sp);
}

// ---------------------------------------------------------------------------
// Weight prep: row-major f32 [K,N] -> bf16 WMMA B-layout tiles.
// Tile (kc,nt) stored as 512 bf16, element index = lane*16 + e,
// with N = nt*16 + lane%16, K = kc*32 + ((e<8) ? e+8*(lane/16) : e+8+8*(lane/16)).
// ---------------------------------------------------------------------------
__global__ void prep_b_kernel(const float* __restrict__ W, int Kdim, int Ndim,
                              int ntCount, __bf16* __restrict__ out) {
    int t = threadIdx.x;                  // 0..511
    int tile = blockIdx.x;                // kc*ntCount + nt
    int nt = tile % ntCount;
    int kc = tile / ntCount;
    int lane = t >> 4, e = t & 15;
    int n = nt * 16 + (lane & 15);
    int k = kc * 32 + ((e < 8) ? (e + 8 * (lane >> 4)) : (e + 8 + 8 * (lane >> 4)));
    float v = (k < Kdim && n < Ndim) ? W[(size_t)k * Ndim + n] : 0.f;
    out[(size_t)tile * 512 + t] = (__bf16)v;
}

// Conv weights (Cout, Cin*9) -> bf16 WMMA A-layout tiles (M=co, K=ci*9+tap).
// Tile (ct,kc) element [lane*16+e]: M = lane%16,
// K = e+16*half  (half = lane/16).
__global__ void prep_convA_kernel(const float* __restrict__ W, int Cin, int Cout,
                                  int kcCount, __bf16* __restrict__ out) {
    int t = threadIdx.x;                  // 0..511
    int tile = blockIdx.x;                // ct*kcCount + kc
    int kc = tile % kcCount;
    int ct = tile / kcCount;
    int lane = t >> 4, e = t & 15;
    int half = lane >> 4;
    int M = lane & 15;
    int K = e + 16 * half;
    int k = kc * 32 + K;
    int co = ct * 16 + M;
    int Kdim = Cin * 9;
    float v = (k < Kdim && co < Cout) ? W[(size_t)co * Kdim + k] : 0.f;
    out[(size_t)tile * 512 + lane * 16 + e] = (__bf16)v;
}

// ---------------------------------------------------------------------------
// Generic bf16 WMMA GEMM: one wave per 16x16 C tile.
// A: row-major bf16 [M, lda] (M multiple of 16). B: pre-tiled (prep_b).
// mode 0: C[m,n] = acc + bias[n]
// mode 1: C[m,n] = mish(acc + bias[n])
// mode 2: atomicAdd(agg[dst[m]*128+n], mish(hg[src[m]*128+n] + acc + bias[n]))
// ---------------------------------------------------------------------------
__global__ void gemm_wmma_kernel(const __bf16* __restrict__ A, int lda,
                                 const __bf16* __restrict__ Bt,
                                 int kcCount, int ntCount, int Mtiles,
                                 const float* __restrict__ bias, int mode,
                                 float* __restrict__ C, int ldc, int nvalid,
                                 const int* __restrict__ src,
                                 const int* __restrict__ dst,
                                 const float* __restrict__ hg,
                                 float* __restrict__ agg) {
    int wid = blockIdx.x * 8 + (threadIdx.x >> 5);
    if (wid >= Mtiles * ntCount) return;
    int lane = threadIdx.x & 31;
    int half = lane >> 4;
    int mt = wid / ntCount;
    int nt = wid - mt * ntCount;

    v8f acc = {0.f, 0.f, 0.f, 0.f, 0.f, 0.f, 0.f, 0.f};
    const __bf16* arow = A + (size_t)(mt * 16 + (lane & 15)) * lda + (half << 3);
    const __bf16* btile = Bt + ((size_t)nt << 9) + (lane << 4);

    for (int kc = 0; kc < kcCount; ++kc) {
        v8bf a0 = *(const v8bf*)(arow + (kc << 5));
        v8bf a1 = *(const v8bf*)(arow + (kc << 5) + 16);
        v16bf av = __builtin_shufflevector(a0, a1, 0, 1, 2, 3, 4, 5, 6, 7,
                                           8, 9, 10, 11, 12, 13, 14, 15);
        v16bf bv = *(const v16bf*)(btile + (((size_t)kc * ntCount) << 9));
        acc = __builtin_amdgcn_wmma_f32_16x16x32_bf16(false, av, false, bv,
                                                      (short)0, acc, false, false);
    }

    int n = nt * 16 + (lane & 15);
    if (n >= nvalid) return;
    float bb = bias[n];
    if (mode == 2) {
#pragma unroll
        for (int v = 0; v < 8; ++v) {
            int m = mt * 16 + v + 8 * half;
            float val = mishf(hg[(size_t)src[m] * 128 + n] + acc[v] + bb);
            atomicAdd(&agg[(size_t)dst[m] * 128 + n], val);
        }
    } else {
#pragma unroll
        for (int v = 0; v < 8; ++v) {
            int m = mt * 16 + v + 8 * half;
            float val = acc[v] + bb;
            if (mode == 1) val = mishf(val);
            C[(size_t)m * ldc + n] = val;
        }
    }
}

// ---------------------------------------------------------------------------
// Node pre-processing: mapping MLP + affine + attr encoder (Linear-LN-ReLU,
// Linear-LN-Mish). One thread per node, weights staged in LDS.
// ---------------------------------------------------------------------------
__global__ void node_pre_kernel(const float* __restrict__ X,
                                const float* __restrict__ w1, const float* __restrict__ b1,
                                const float* __restrict__ w2, const float* __restrict__ b2,
                                const float* __restrict__ gam, const float* __restrict__ bet,
                                const float* __restrict__ n1w, const float* __restrict__ n1b,
                                const float* __restrict__ n2w, const float* __restrict__ n2b,
                                float* __restrict__ hF, __bf16* __restrict__ Abuf) {
    __shared__ float sW1[2560], sW2[2560], sN2[4096], sN1[640], sB1[256];
    __shared__ float sB2[10], sG[10], sBt[10], sN1b[64], sN2b[64];
    int tid = threadIdx.x;
    for (int i = tid; i < 2560; i += 256) { sW1[i] = w1[i]; sW2[i] = w2[i]; }
    for (int i = tid; i < 4096; i += 256) sN2[i] = n2w[i];
    for (int i = tid; i < 640; i += 256) sN1[i] = n1w[i];
    sB1[tid] = b1[tid];
    if (tid < 10) { sB2[tid] = b2[tid]; sG[tid] = gam[tid]; sBt[tid] = bet[tid]; }
    if (tid < 64) { sN1b[tid] = n1b[tid]; sN2b[tid] = n2b[tid]; }
    __syncthreads();

    int n = blockIdx.x * 256 + tid;
    if (n >= N_NODES) return;

    float x[10], y[10];
#pragma unroll
    for (int k = 0; k < 10; k++) { x[k] = X[(size_t)n * 10 + k]; y[k] = sB2[k]; }
    for (int j = 0; j < 256; j++) {
        float t = sB1[j];
#pragma unroll
        for (int k = 0; k < 10; k++) t += x[k] * sW1[k * 256 + j];
        t = fmaxf(t, 0.f);
#pragma unroll
        for (int k = 0; k < 10; k++) y[k] += t * sW2[j * 10 + k];
    }
    float ma[10];
#pragma unroll
    for (int k = 0; k < 10; k++) ma[k] = sG[k] * (x[k] + y[k]) + sBt[k];

    float a[64];
    float s1 = 0.f, s2 = 0.f;
#pragma unroll
    for (int j = 0; j < 64; j++) {
        float t = sN1b[j];
#pragma unroll
        for (int k = 0; k < 10; k++) t += ma[k] * sN1[k * 64 + j];
        a[j] = t; s1 += t; s2 += t * t;
    }
    float m = s1 * (1.f / 64.f);
    float var = s2 * (1.f / 64.f) - m * m;
    float inv = rsqrtf(var + 1e-5f);
#pragma unroll
    for (int j = 0; j < 64; j++) a[j] = fmaxf((a[j] - m) * inv, 0.f);

    float c[64];
    s1 = 0.f; s2 = 0.f;
#pragma unroll
    for (int j = 0; j < 64; j++) {
        float t = sN2b[j];
#pragma unroll
        for (int k = 0; k < 64; k++) t += a[k] * sN2[k * 64 + j];
        c[j] = t; s1 += t; s2 += t * t;
    }
    m = s1 * (1.f / 64.f);
    var = s2 * (1.f / 64.f) - m * m;
    inv = rsqrtf(var + 1e-5f);
#pragma unroll
    for (int j = 0; j < 64; j++) {
        float t = mishf((c[j] - m) * inv);
        hF[(size_t)n * 128 + j] = t;
        Abuf[(size_t)n * 256 + j] = (__bf16)t;
    }
}

// ---------------------------------------------------------------------------
// Edge encoder: Linear-LN-ReLU, Linear-LN-Mish -> bf16 e_feat [E,64]
// ---------------------------------------------------------------------------
__global__ void edge_pre_kernel(const float* __restrict__ EX,
                                const float* __restrict__ e1w, const float* __restrict__ e1b,
                                const float* __restrict__ e2w, const float* __restrict__ e2b,
                                __bf16* __restrict__ efeat) {
    __shared__ float sW1[768], sW2[4096], sB1[64], sB2[64];
    int tid = threadIdx.x;
    for (int i = tid; i < 768; i += 256) sW1[i] = e1w[i];
    for (int i = tid; i < 4096; i += 256) sW2[i] = e2w[i];
    if (tid < 64) { sB1[tid] = e1b[tid]; sB2[tid] = e2b[tid]; }
    __syncthreads();

    int n = blockIdx.x * 256 + tid;
    if (n >= N_EDGES) return;

    float x[12];
#pragma unroll
    for (int k = 0; k < 12; k++) x[k] = EX[(size_t)n * 12 + k];
    float a[64];
    float s1 = 0.f, s2 = 0.f;
#pragma unroll
    for (int j = 0; j < 64; j++) {
        float t = sB1[j];
#pragma unroll
        for (int k = 0; k < 12; k++) t += x[k] * sW1[k * 64 + j];
        a[j] = t; s1 += t; s2 += t * t;
    }
    float m = s1 * (1.f / 64.f);
    float var = s2 * (1.f / 64.f) - m * m;
    float inv = rsqrtf(var + 1e-5f);
#pragma unroll
    for (int j = 0; j < 64; j++) a[j] = fmaxf((a[j] - m) * inv, 0.f);

    float c[64];
    s1 = 0.f; s2 = 0.f;
#pragma unroll
    for (int j = 0; j < 64; j++) {
        float t = sB2[j];
#pragma unroll
        for (int k = 0; k < 64; k++) t += a[k] * sW2[k * 64 + j];
        c[j] = t; s1 += t; s2 += t * t;
    }
    m = s1 * (1.f / 64.f);
    var = s2 * (1.f / 64.f) - m * m;
    inv = rsqrtf(var + 1e-5f);
#pragma unroll
    for (int j = 0; j < 64; j++)
        efeat[(size_t)n * 64 + j] = (__bf16)mishf((c[j] - m) * inv);
}

// ---------------------------------------------------------------------------
// Conv tower: one block per node; padded [C][12][12] bf16 images in LDS;
// each conv is a 9-tap implicit GEMM done with bf16 WMMA.
// ---------------------------------------------------------------------------
__device__ void conv_pass(const __bf16* __restrict__ actIn, __bf16* __restrict__ actOut,
                          int Cin, int Cout, const __bf16* __restrict__ At,
                          const float* __restrict__ bias, int kcCount,
                          int* sBK, float* gsum, bool pool, int tid) {
    int Kdim = Cin * 9;
    for (int k = tid; k < kcCount * 32; k += 256) {
        int kk = (k < Kdim) ? k : 0;
        int ci = kk / 9; int tap = kk - ci * 9;
        int dy = tap / 3; int dx = tap - dy * 3;
        sBK[k] = ci * 144 + (dy - 1) * 12 + (dx - 1);
    }
    if (pool) {
        if (tid < 64) gsum[tid] = 0.f;
    } else {
        for (int i = tid; i < Cout * 144; i += 256) actOut[i] = (__bf16)0.f;
    }
    __syncthreads();

    int wave = tid >> 5, lane = tid & 31, half = lane >> 4;
    int cotiles = Cout >> 4;
    if (wave < 7) {                                 // 7 pixel tiles cover 112 >= 100 px
        int p = wave * 16 + (lane & 15);
        int y = p / 10, xx = p - y * 10;
        int offp = (p < 100) ? ((y + 1) * 12 + (xx + 1)) : 13;
        v8f zz = {0.f, 0.f, 0.f, 0.f, 0.f, 0.f, 0.f, 0.f};
        v8f acc[4]; acc[0] = zz; acc[1] = zz; acc[2] = zz; acc[3] = zz;

        for (int kc = 0; kc < kcCount; ++kc) {
            v16bf bv;
            int kbase = kc * 32 + 16 * half;
#pragma unroll
            for (int e = 0; e < 16; ++e) {
                int kg = kbase + e;
                bv[e] = (kg < Kdim) ? actIn[sBK[kg] + offp] : (__bf16)0.f;
            }
#pragma unroll
            for (int ct = 0; ct < 4; ++ct) {
                if (ct < cotiles) {
                    v16bf av = *(const v16bf*)(At + (((size_t)(ct * kcCount + kc)) << 9)
                                               + (lane << 4));
                    acc[ct] = __builtin_amdgcn_wmma_f32_16x16x32_bf16(
                        false, av, false, bv, (short)0, acc[ct], false, false);
                }
            }
        }
        if (p < 100) {
#pragma unroll
            for (int ct = 0; ct < 4; ++ct) {
                if (ct < cotiles) {
#pragma unroll
                    for (int v = 0; v < 8; ++v) {
                        int co = ct * 16 + v + 8 * half;
                        float val = mishf(acc[ct][v] + bias[co]);
                        if (pool) atomicAdd(&gsum[co], val);
                        else actOut[co * 144 + offp] = (__bf16)val;
                    }
                }
            }
        }
    }
    __syncthreads();
}

__global__ void grid_conv_kernel(const float* __restrict__ G,
                                 const __bf16* __restrict__ At,
                                 const float* __restrict__ b1,
                                 const float* __restrict__ b2,
                                 const float* __restrict__ b3,
                                 float* __restrict__ hF, __bf16* __restrict__ Abuf) {
    __shared__ __bf16 sA[32 * 144];
    __shared__ __bf16 sB[32 * 144];
    __shared__ int sBK[320];
    __shared__ float sG[64];
    int tid = threadIdx.x;
    int node = blockIdx.x;

    for (int i = tid; i < 7 * 144; i += 256) sA[i] = (__bf16)0.f;
    __syncthreads();
    const float* gp = G + (size_t)node * 700;
    for (int i = tid; i < 700; i += 256) {
        int c = i / 100, p = i - c * 100, y = p / 10, x = p - y * 10;
        sA[c * 144 + (y + 1) * 12 + (x + 1)] = (__bf16)gp[i];
    }
    __syncthreads();

    conv_pass(sA, sB, 7, 16, At, b1, 2, sBK, sG, false, tid);
    conv_pass(sB, sA, 16, 32, At + 1024, b2, 5, sBK, sG, false, tid);
    conv_pass(sA, sB, 32, 64, At + 6144, b3, 9, sBK, sG, true, tid);

    if (tid < 64) {
        float g = sG[tid] * 0.01f;                  // mean over 100 pixels
        hF[(size_t)node * 128 + 64 + tid] = g;
        Abuf[(size_t)node * 256 + 64 + tid] = (__bf16)g;
    }
}

// ---------------------------------------------------------------------------
// Misc elementwise / reduction kernels
// ---------------------------------------------------------------------------
__global__ void agg_to_abuf_kernel(const float* __restrict__ agg, __bf16* __restrict__ Abuf) {
    size_t i = (size_t)blockIdx.x * 256 + threadIdx.x;   // 50000*128 exact
    size_t m = i >> 7; int c = (int)(i & 127);
    Abuf[m * 256 + 128 + c] = (__bf16)agg[i];
}

__global__ void ln_update_kernel(float* __restrict__ h, const float* __restrict__ upd,
                                 __bf16* __restrict__ Abuf) {
    __shared__ float rs[128], rq[128];
    int r = blockIdx.x, c = threadIdx.x;
    size_t idx = (size_t)r * 128 + c;
    float s = h[idx] + upd[idx];
    rs[c] = s; rq[c] = s * s;
    __syncthreads();
    for (int st = 64; st > 0; st >>= 1) {
        if (c < st) { rs[c] += rs[c + st]; rq[c] += rq[c + st]; }
        __syncthreads();
    }
    float m = rs[0] * (1.f / 128.f);
    float var = rq[0] * (1.f / 128.f) - m * m;
    float hn = (s - m) * rsqrtf(var + 1e-5f);
    h[idx] = hn;
    Abuf[(size_t)r * 256 + c] = (__bf16)hn;
}

__global__ void pool_acc_kernel(const float* __restrict__ h, const int* __restrict__ gid,
                                float* __restrict__ gsum, float* __restrict__ gcnt) {
    int r = blockIdx.x, c = threadIdx.x;
    int g = gid[r];
    atomicAdd(&gsum[(size_t)g * 128 + c], h[(size_t)r * 128 + c]);
    if (c == 0) atomicAdd(&gcnt[g], 1.f);
}

__global__ void head_a_kernel(const float* __restrict__ gsum, const float* __restrict__ gcnt,
                              const int* __restrict__ gid, __bf16* __restrict__ Abuf) {
    int r = blockIdx.x, c = threadIdx.x;
    int g = gid[r];
    float v = gsum[(size_t)g * 128 + c] / gcnt[g];
    Abuf[(size_t)r * 256 + 128 + c] = (__bf16)v;
}

__global__ void ln_mish_head_kernel(const float* __restrict__ z0, __bf16* __restrict__ Abuf) {
    __shared__ float rs[256], rq[256];
    int r = blockIdx.x, c = threadIdx.x;
    size_t idx = (size_t)r * 256 + c;
    float s = z0[idx];
    rs[c] = s; rq[c] = s * s;
    __syncthreads();
    for (int st = 128; st > 0; st >>= 1) {
        if (c < st) { rs[c] += rs[c + st]; rq[c] += rq[c + st]; }
        __syncthreads();
    }
    float m = rs[0] * (1.f / 256.f);
    float var = rq[0] * (1.f / 256.f) - m * m;
    float z = mishf((s - m) * rsqrtf(var + 1e-5f));
    Abuf[idx] = (__bf16)z;
}

// ---------------------------------------------------------------------------
// Driver
// ---------------------------------------------------------------------------
extern "C" void kernel_launch(void* const* d_in, const int* in_sizes, int n_in,
                              void* d_out, int out_size, void* d_ws, size_t ws_size,
                              hipStream_t stream) {
    (void)in_sizes; (void)n_in; (void)out_size; (void)ws_size;

    const float* node_x    = (const float*)d_in[0];
    const float* node_grid = (const float*)d_in[1];
    const float* edge_x    = (const float*)d_in[2];
    const int*   eidx      = (const int*)d_in[3];
    const int*   gid       = (const int*)d_in[4];
    const float* ma1w = (const float*)d_in[5],  *ma1b = (const float*)d_in[6];
    const float* ma2w = (const float*)d_in[7],  *ma2b = (const float*)d_in[8];
    const float* gam  = (const float*)d_in[9],  *bet  = (const float*)d_in[10];
    const float* n1w  = (const float*)d_in[11], *n1b  = (const float*)d_in[12];
    const float* n2w  = (const float*)d_in[13], *n2b  = (const float*)d_in[14];
    const float* g1w  = (const float*)d_in[15], *g1b  = (const float*)d_in[16];
    const float* g2w  = (const float*)d_in[17], *g2b  = (const float*)d_in[18];
    const float* g3w  = (const float*)d_in[19], *g3b  = (const float*)d_in[20];
    const float* e1w  = (const float*)d_in[21], *e1b  = (const float*)d_in[22];
    const float* e2w  = (const float*)d_in[23], *e2b  = (const float*)d_in[24];
    const float* wew[3]; const float* web[3]; const float* wnw[3]; const float* wnb[3];
    for (int l = 0; l < 3; l++) {
        wew[l] = (const float*)d_in[25 + 4 * l];
        web[l] = (const float*)d_in[26 + 4 * l];
        wnw[l] = (const float*)d_in[27 + 4 * l];
        wnb[l] = (const float*)d_in[28 + 4 * l];
    }
    const float* h1w = (const float*)d_in[37], *h1b = (const float*)d_in[38];
    const float* h2w = (const float*)d_in[39], *h2b = (const float*)d_in[40];

    char* ws = (char*)d_ws;
    __bf16* convA = (__bf16*)(ws);                       // 48 KiB of A tiles
    __bf16* BT    = (__bf16*)(ws + 65536);               // B tiles region
    __bf16* weB[3]; __bf16* wnB[3];
    for (int l = 0; l < 3; l++) { weB[l] = BT + l * 8192; wnB[l] = BT + 24576 + l * 32768; }
    __bf16* h1B = BT + 122880;
    __bf16* h2B = BT + 188416;
    float* gsum = (float*)(ws + 524288);                 // [250,128]
    float* gcnt = gsum + 32000;                          // [250]
    __bf16* efeat = (__bf16*)(ws + 1048576);             // [400000,64] bf16 (51.2 MB)
    float*  z0    = (float*)(ws + 1048576);              // reuse as [50000,256] f32
    __bf16* Abuf  = (__bf16*)(ws + 52248576);            // [50000,256] bf16
    float*  agg   = (float*)(ws + 77848576);             // [50000,128] f32
    float*  upd   = (float*)(ws + 103448576);            // [50000,128] f32

    float* seg = (float*)d_out;                          // [50000,25]
    float* hF  = seg + (size_t)N_NODES * 25;             // [50000,128] final h lives here
    const int* srcI = eidx;
    const int* dstI = eidx + N_EDGES;

    // --- weight layout prep ---
    prep_convA_kernel<<<2,  512, 0, stream>>>(g1w, 7, 16, 2, convA);
    prep_convA_kernel<<<10, 512, 0, stream>>>(g2w, 16, 32, 5, convA + 1024);
    prep_convA_kernel<<<36, 512, 0, stream>>>(g3w, 32, 64, 9, convA + 6144);
    for (int l = 0; l < 3; l++) {
        prep_b_kernel<<<16, 512, 0, stream>>>(wew[l], 64, 128, 8, weB[l]);
        prep_b_kernel<<<64, 512, 0, stream>>>(wnw[l], 256, 128, 8, wnB[l]);
    }
    prep_b_kernel<<<128, 512, 0, stream>>>(h1w, 256, 256, 16, h1B);
    prep_b_kernel<<<16,  512, 0, stream>>>(h2w, 256, 25, 2, h2B);

    // --- encoders ---
    node_pre_kernel<<<(N_NODES + 255) / 256, 256, 0, stream>>>(
        node_x, ma1w, ma1b, ma2w, ma2b, gam, bet, n1w, n1b, n2w, n2b, hF, Abuf);
    grid_conv_kernel<<<N_NODES, 256, 0, stream>>>(node_grid, convA, g1b, g2b, g3b, hF, Abuf);
    edge_pre_kernel<<<(N_EDGES + 255) / 256, 256, 0, stream>>>(edge_x, e1w, e1b, e2w, e2b, efeat);

    // --- GNN layers ---
    for (int l = 0; l < 3; l++) {
        hipMemsetAsync(agg, 0, (size_t)N_NODES * 128 * sizeof(float), stream);
        // ef = e@we+b fused with msg = mish(h[src]+ef) and segment-sum into agg
        gemm_wmma_kernel<<<25000, 256, 0, stream>>>(efeat, 64, weB[l], 2, 8, 25000,
                                                    web[l], 2, nullptr, 0, 128,
                                                    srcI, dstI, hF, agg);
        agg_to_abuf_kernel<<<25000, 256, 0, stream>>>(agg, Abuf);
        // upd = mish(concat(h,agg)@wn + b)
        gemm_wmma_kernel<<<3125, 256, 0, stream>>>(Abuf, 256, wnB[l], 8, 8, 3125,
                                                   wnb[l], 1, upd, 128, 128,
                                                   nullptr, nullptr, nullptr, nullptr);
        ln_update_kernel<<<N_NODES, 128, 0, stream>>>(hF, upd, Abuf);
    }

    // --- graph pooling + head ---
    hipMemsetAsync(gsum, 0, (size_t)(32000 + 250) * sizeof(float), stream);
    pool_acc_kernel<<<N_NODES, 128, 0, stream>>>(hF, gid, gsum, gcnt);
    head_a_kernel<<<N_NODES, 128, 0, stream>>>(gsum, gcnt, gid, Abuf);

    gemm_wmma_kernel<<<6250, 256, 0, stream>>>(Abuf, 256, h1B, 8, 16, 3125,
                                               h1b, 0, z0, 256, 256,
                                               nullptr, nullptr, nullptr, nullptr);
    ln_mish_head_kernel<<<N_NODES, 256, 0, stream>>>(z0, Abuf);
    gemm_wmma_kernel<<<782, 256, 0, stream>>>(Abuf, 256, h2B, 8, 2, 3125,
                                              h2b, 0, seg, 25, 25,
                                              nullptr, nullptr, nullptr, nullptr);
}
